// AdditiveAttention_45363444580675
// MI455X (gfx1250) — matrix operand order repaired and checked by reading, not compiled, not run on hardware
//
#include <hip/hip_runtime.h>
#include <hip/hip_bf16.h>

// ---- problem constants (match reference) ----
#define BB 8
#define QQ 256
#define KK 256
#define HH 256
#define DV 256

typedef __attribute__((ext_vector_type(16))) _Float16 v16h;
typedef __attribute__((ext_vector_type(8)))  _Float16 v8h;
typedef __attribute__((ext_vector_type(8)))  float    v8f;

#if defined(__has_builtin)
# if __has_builtin(__builtin_amdgcn_tanhf)
#   define TANHF(x) __builtin_amdgcn_tanhf(x)
# endif
#endif
#ifndef TANHF
# define TANHF(x) tanhf(x)
#endif

// ---------------- fp32 -> f16 copy ----------------
__global__ void cvt_f16_kernel(const float* __restrict__ src,
                               _Float16* __restrict__ dst, int n) {
  int i = blockIdx.x * blockDim.x + threadIdx.x;
  if (i < n) dst[i] = (_Float16)src[i];
}

// ------------- fp32 (R x C) -> f16 transposed (C x R), batched over z -------------
__global__ void transpose_cvt_f16_kernel(const float* __restrict__ src,
                                         _Float16* __restrict__ dst,
                                         int R, int C) {
  size_t base = (size_t)blockIdx.z * R * C;
  int i = blockIdx.x * blockDim.x + threadIdx.x;
  if (i < R * C) {
    int r = i / C, c = i % C;
    dst[base + (size_t)c * R + r] = (_Float16)src[base + (size_t)r * C + c];
  }
}

// ---------------- WMMA f16 GEMM: C(MxN,f32) = A(MxK,f16 row-major) x Bt(NxK,f16 row-major) ----------------
// One wave computes one 16x16 tile; blockDim=(32,4) -> 4 N-subtiles per block.
__global__ __launch_bounds__(128)
void gemm16_wmma_kernel(const _Float16* __restrict__ A,
                        const _Float16* __restrict__ Bt,
                        float* __restrict__ C,
                        int M, int N, int Kd,
                        long long sA, long long sB, long long sC) {
  A  += (size_t)blockIdx.z * sA;
  Bt += (size_t)blockIdx.z * sB;
  C  += (size_t)blockIdx.z * sC;

  const int lane = threadIdx.x;           // 0..31 (wave32)
  const int m0 = blockIdx.x * 16;
  const int n0 = (blockIdx.y * 4 + threadIdx.y) * 16;
  if (m0 >= M || n0 >= N) return;         // wave-uniform

  v8f acc = {};

  const int r15   = lane & 15;
  const int hiHalf = lane >> 4;           // 0 for lanes 0-15, 1 for 16-31
  // A fragment per ISA: lanes<16 hold K {0..7,16..23}; lanes>=16 hold {8..15,24..31}
  const int aKoff = hiHalf << 3;          // 0 or 8
  // B fragment matches A: lanes<16 hold K {0..7,16..23}; lanes>=16 hold {8..15,24..31}
  const int bKoff = hiHalf << 3;          // 0 or 8

  const _Float16* aRow = A  + (size_t)(m0 + r15) * Kd;
  const _Float16* bRow = Bt + (size_t)(n0 + r15) * Kd;

  for (int k0 = 0; k0 < Kd; k0 += 32) {
    union { v16h v; v8h h[2]; } ua, ub;
    ua.h[0] = *(const v8h*)(aRow + k0 + aKoff);
    ua.h[1] = *(const v8h*)(aRow + k0 + aKoff + 16);
    ub.h[0] = *(const v8h*)(bRow + k0 + bKoff);
    ub.h[1] = *(const v8h*)(bRow + k0 + bKoff + 16);
    acc = __builtin_amdgcn_wmma_f32_16x16x32_f16(
        /*neg_a=*/false, ua.v, /*neg_b=*/false, ub.v,
        /*c_mod=*/(short)0, acc, /*reuse_a=*/false, /*reuse_b=*/false);
  }

  // D layout: lanes<16: N=lane, M=m0+v ; lanes>=16: N=lane-16, M=m0+8+v
  const int n  = n0 + r15;
  const int mB = m0 + (hiHalf << 3);
#pragma unroll
  for (int v = 0; v < 8; ++v)
    C[(size_t)(mB + v) * N + n] = acc[v];
}

// ---------------- fused score (tanh) + mask + softmax ----------------
// One block = one (b, 16-row q tile). 256 threads; thread t owns key column k=t.
__global__ __launch_bounds__(256)
void score_softmax_kernel(const float* __restrict__ qf,   // (B*Q, H)
                          const float* __restrict__ kf,   // (B*K, H)
                          const float* __restrict__ wv,   // (H)
                          const int*   __restrict__ vlens,// (B)
                          _Float16* __restrict__ attn) {  // (B*Q, K) f16
  __shared__ __align__(16) float sQ[16 * HH];   // qf tile  [q][h]
  __shared__ __align__(16) float sW[HH];        // w_v
  __shared__ __align__(16) float sS[16 * KK];   // scores   [q][k]

  const int tid = threadIdx.x;                  // 0..255 == k index
  const int b   = blockIdx.x / (QQ / 16);
  const int q0  = (blockIdx.x % (QQ / 16)) * 16;

  sW[tid] = wv[tid];
  {
    const int q = tid >> 4;
    const int h = (tid & 15) * 16;
    const float* src = qf + ((size_t)(b * QQ + q0 + q)) * HH + h;
    float* dst = &sQ[q * HH + h];
#pragma unroll
    for (int j = 0; j < 4; ++j)
      *(float4*)(dst + 4 * j) = *(const float4*)(src + 4 * j);
  }
  __syncthreads();

  float acc[16];
#pragma unroll
  for (int q = 0; q < 16; ++q) acc[q] = 0.f;

  const float* krow = kf + ((size_t)(b * KK + tid)) * HH;
  for (int h0 = 0; h0 < HH; h0 += 4) {
    const float4 kv  = *(const float4*)(krow + h0);
    const float4 wv4 = *(const float4*)(&sW[h0]);
#pragma unroll
    for (int q = 0; q < 16; ++q) {
      const float4 qv = *(const float4*)(&sQ[q * HH + h0]);
      acc[q] += wv4.x * TANHF(qv.x + kv.x)
              + wv4.y * TANHF(qv.y + kv.y)
              + wv4.z * TANHF(qv.z + kv.z)
              + wv4.w * TANHF(qv.w + kv.w);
    }
  }

  const bool masked = (tid >= vlens[b]);
#pragma unroll
  for (int q = 0; q < 16; ++q)
    sS[q * KK + tid] = masked ? -1000000.0f : acc[q];
  __syncthreads();

  // softmax over k: 8 waves x 2 rows each; lane owns 8 strided elements
  const int wave = tid >> 5, lane = tid & 31;
#pragma unroll
  for (int rr = 0; rr < 2; ++rr) {
    const int r = wave * 2 + rr;
    float vals[8], m = -INFINITY;
#pragma unroll
    for (int j = 0; j < 8; ++j) {
      vals[j] = sS[r * KK + lane + 32 * j];
      m = fmaxf(m, vals[j]);
    }
#pragma unroll
    for (int off = 16; off >= 1; off >>= 1)
      m = fmaxf(m, __shfl_xor(m, off, 32));
    float sum = 0.f;
#pragma unroll
    for (int j = 0; j < 8; ++j) { vals[j] = __expf(vals[j] - m); sum += vals[j]; }
#pragma unroll
    for (int off = 16; off >= 1; off >>= 1)
      sum += __shfl_xor(sum, off, 32);
    const float inv = 1.0f / sum;
    _Float16* orow = attn + ((size_t)(b * QQ + q0 + r)) * KK;
#pragma unroll
    for (int j = 0; j < 8; ++j)
      orow[lane + 32 * j] = (_Float16)(vals[j] * inv);
  }
}

// ---------------- host-side orchestration ----------------
static inline size_t align256(size_t x) { return (x + 255) & ~(size_t)255; }

extern "C" void kernel_launch(void* const* d_in, const int* in_sizes, int n_in,
                              void* d_out, int out_size, void* d_ws, size_t ws_size,
                              hipStream_t stream) {
  const float* queries = (const float*)d_in[0];  // (B,Q,DQ)
  const float* keys    = (const float*)d_in[1];  // (B,K,DK)
  const float* values  = (const float*)d_in[2];  // (B,K,DV)
  const int*   vlens   = (const int*)  d_in[3];  // (B)
  const float* W_q     = (const float*)d_in[4];  // (DQ,H)
  const float* W_k     = (const float*)d_in[5];  // (DK,H)
  const float* w_v     = (const float*)d_in[6];  // (H)
  float* out = (float*)d_out;                    // (B,Q,DV) f32

  // workspace carve-up
  char* w = (char*)d_ws;
  size_t off = 0;
  auto take = [&](size_t bytes) { void* p = w + off; off = align256(off + bytes); return p; };
  _Float16* qh   = (_Float16*)take((size_t)BB * QQ * HH * 2); // queries f16
  _Float16* kh   = (_Float16*)take((size_t)BB * KK * HH * 2); // keys f16
  _Float16* WqT  = (_Float16*)take((size_t)HH * HH * 2);      // W_q^T f16 (out,in)
  _Float16* WkT  = (_Float16*)take((size_t)HH * HH * 2);
  _Float16* vT   = (_Float16*)take((size_t)BB * DV * KK * 2); // values^T f16 (b,d,k)
  float*    qf   = (float*)   take((size_t)BB * QQ * HH * 4);
  float*    kf   = (float*)   take((size_t)BB * KK * HH * 4);
  _Float16* attn = (_Float16*)take((size_t)BB * QQ * KK * 2);
  (void)ws_size;

  // 1) f16 conversions of queries/keys
  {
    int n = BB * QQ * HH;
    cvt_f16_kernel<<<(n + 255) / 256, 256, 0, stream>>>(queries, qh, n);
    cvt_f16_kernel<<<(n + 255) / 256, 256, 0, stream>>>(keys, kh, n);
  }
  // 2) transposed f16 weights and values
  {
    int n = HH * HH;
    transpose_cvt_f16_kernel<<<dim3((n + 255) / 256, 1, 1), 256, 0, stream>>>(W_q, WqT, HH, HH);
    transpose_cvt_f16_kernel<<<dim3((n + 255) / 256, 1, 1), 256, 0, stream>>>(W_k, WkT, HH, HH);
    int nv = KK * DV;
    transpose_cvt_f16_kernel<<<dim3((nv + 255) / 256, 1, BB), 256, 0, stream>>>(values, vT, KK, DV);
  }
  // 3) projections: qf = qh @ W_q, kf = kh @ W_k   (M = B*256, N = 256, K = 256)
  {
    dim3 grid(BB * QQ / 16, HH / 64, 1), block(32, 4, 1);
    gemm16_wmma_kernel<<<grid, block, 0, stream>>>(qh, WqT, qf, BB * QQ, HH, HH, 0, 0, 0);
    gemm16_wmma_kernel<<<grid, block, 0, stream>>>(kh, WkT, kf, BB * KK, HH, HH, 0, 0, 0);
  }
  // 4) tanh scores + mask + softmax -> attn (f16)
  score_softmax_kernel<<<BB * (QQ / 16), 256, 0, stream>>>(qf, kf, w_v, vlens, attn);
  // 5) out[b] = attn[b] @ values[b]  (batched, M=N=K=256)
  {
    dim3 grid(QQ / 16, DV / 64, BB), block(32, 4, 1);
    gemm16_wmma_kernel<<<grid, block, 0, stream>>>(
        attn, vT, out, QQ, DV, KK,
        (long long)QQ * KK, (long long)DV * KK, (long long)QQ * DV);
  }
}
